// SlotAttentionModule_43877385896079
// MI455X (gfx1250) — matrix operand order repaired and checked by reading, not compiled, not run on hardware
//
#include <hip/hip_runtime.h>
#include <hip/hip_bf16.h>
#include <math.h>

// ---------------------------------------------------------------------------
// Slot Attention for MI455X (gfx1250, wave32, WMMA f16->f32)
//   B=256, N=4096, D=64, K=16, 3 iters.  Memory-bound: k/v kept in f16,
//   attn never materialized for iterations (eps-renorm folded into U = p@v).
//   k computed as wk@x^T so its C/D fragment stores contiguously (b128);
//   v computed as x@wv^T so v^T stores contiguously (b128).
// ---------------------------------------------------------------------------

typedef __attribute__((ext_vector_type(16))) _Float16 v16h;
typedef __attribute__((ext_vector_type(8)))  _Float16 v8h;
typedef __attribute__((ext_vector_type(8)))  float    v8f;

#define B_SZ      256
#define N_SZ      4096
#define D_SZ      64
#define K_SL      16
#define N_ITERS   3
#define EPS_A     1e-8f
#define LN_EPS    1e-5f

union V16 { v16h v; v8h h[2]; };

static __device__ __forceinline__ v8f wmma16(const v16h& a, const v16h& b, v8f c) {
  // D = A(16x32 f16) x B(32x16 f16) + C(16x16 f32)
  return __builtin_amdgcn_wmma_f32_16x16x32_f16(false, a, false, b, (short)0, c,
                                                false, false);
}

// ===========================================================================
// Kernel A: x = LN(inputs)*g+b
//   k = wk @ x^T   -> C/D: M=feature, N=position -> contiguous store k_nat[n][d]
//   v = x @ wv^T   -> C/D: M=position, N=feature -> contiguous store v_t[d][n]
// One block = 256 consecutive rows (all in one batch since N%256==0).
// ===========================================================================
__global__ __launch_bounds__(256) void kv_proj_kernel(
    const float* __restrict__ inputs, const float* __restrict__ g_in,
    const float* __restrict__ b_in,
    const float* __restrict__ wk, const float* __restrict__ bk,
    const float* __restrict__ wv, const float* __restrict__ bv,
    _Float16* __restrict__ k_nat, _Float16* __restrict__ v_t) {
  __shared__ _Float16 xln[256][64];

  const int tid  = threadIdx.x;
  const int lane = tid & 31, wave = tid >> 5;
  const int c = lane & 15, hi = lane >> 4;

  // ---- per-thread LayerNorm of one row (64 f32, kept in VGPRs) ----
  const size_t row = (size_t)blockIdx.x * 256 + tid;
  const float4* rp = (const float4*)(inputs + row * 64);
  float x[64];
#pragma unroll
  for (int i = 0; i < 16; ++i) {
    float4 f = rp[i];
    x[4*i+0] = f.x; x[4*i+1] = f.y; x[4*i+2] = f.z; x[4*i+3] = f.w;
  }
  float m = 0.f;
#pragma unroll
  for (int i = 0; i < 64; ++i) m += x[i];
  m *= (1.f / 64.f);
  float var = 0.f;
#pragma unroll
  for (int i = 0; i < 64; ++i) { float d = x[i] - m; var += d * d; }
  var *= (1.f / 64.f);
  const float inv = rsqrtf(var + LN_EPS);
#pragma unroll
  for (int i = 0; i < 64; ++i)
    xln[tid][i] = (_Float16)((x[i] - m) * inv * g_in[i] + b_in[i]);
  __syncthreads();

  const int bb         = blockIdx.x >> 4;          // batch index (16 blocks/batch)
  const int nBlockBase = (blockIdx.x & 15) * 256;  // position base within batch
  const size_t rowBlk  = (size_t)blockIdx.x * 256; // global row base

  // Each wave owns 32 rows/positions: [wave*32, wave*32+32)
  for (int mt = 0; mt < 2; ++mt) {
    const int mbase = wave * 32 + mt * 16;  // local row/pos tile base

    // ================= v = x @ wv^T  (M=rows, N=features) =================
    {
      // A-fragments of x (16x32 f16: lane<16 -> K 0..15; lane>=16 -> K 16..31)
      V16 a0, a1;
      const int r = mbase + c;
      const int koff = hi ? 16 : 0;
      a0.h[0] = *(const v8h*)&xln[r][koff];
      a0.h[1] = *(const v8h*)&xln[r][koff + 8];
      a1.h[0] = *(const v8h*)&xln[r][32 + koff];
      a1.h[1] = *(const v8h*)&xln[r][32 + koff + 8];
#pragma unroll
      for (int j = 0; j < 4; ++j) {
        const float* wvr = wv + (size_t)(j * 16 + c) * 64;
        const int kb2 = hi * 16;  // B layout: lane<16 -> K 0..15; lane>=16 -> 16..31
        V16 b0, b1;
#pragma unroll
        for (int i = 0; i < 8; ++i) {
          b0.h[0][i] = (_Float16)wvr[kb2 + i];
          b0.h[1][i] = (_Float16)wvr[kb2 + 8 + i];
          b1.h[0][i] = (_Float16)wvr[32 + kb2 + i];
          b1.h[1][i] = (_Float16)wvr[32 + kb2 + 8 + i];
        }
        v8f accv;
        const float bvv = bv[j * 16 + c];
#pragma unroll
        for (int g = 0; g < 8; ++g) accv[g] = bvv;
        accv = wmma16(a0.v, b0.v, accv);
        accv = wmma16(a1.v, b1.v, accv);
        // v_t[b][d][n]: lane holds fixed d=j*16+c, g spans consecutive n
        v8h pv;
#pragma unroll
        for (int g = 0; g < 8; ++g) pv[g] = (_Float16)accv[g];
        const int d  = j * 16 + c;
        const int n0 = nBlockBase + mbase + 8 * hi;
        *(v8h*)&v_t[((size_t)bb * 64 + d) * N_SZ + n0] = pv;
      }
    }

    // ================= k = wk @ x^T  (M=features, N=positions) =============
    {
      // B-fragment of x^T: col = position, K = d (contiguous in LDS row)
      V16 xb0, xb1;
      const int prow = mbase + c;
      const int kb2 = hi * 8;
      xb0.h[0] = *(const v8h*)&xln[prow][kb2];
      xb0.h[1] = *(const v8h*)&xln[prow][kb2 + 16];
      xb1.h[0] = *(const v8h*)&xln[prow][32 + kb2];
      xb1.h[1] = *(const v8h*)&xln[prow][32 + kb2 + 16];
#pragma unroll
      for (int ft = 0; ft < 4; ++ft) {
        // A-fragment of wk: row = feature ft*16 + (lane&15), interleaved K
        const float* wkr = wk + (size_t)(ft * 16 + c) * 64;
        const int koff = hi ? 8 : 0;
        V16 a0, a1;
#pragma unroll
        for (int i = 0; i < 8; ++i) {
          a0.h[0][i] = (_Float16)wkr[koff + i];
          a0.h[1][i] = (_Float16)wkr[koff + 16 + i];
          a1.h[0][i] = (_Float16)wkr[32 + koff + i];
          a1.h[1][i] = (_Float16)wkr[32 + koff + 16 + i];
        }
        v8f acck;
        const float4 bk0 = *(const float4*)&bk[ft * 16 + 8 * hi];
        const float4 bk1 = *(const float4*)&bk[ft * 16 + 8 * hi + 4];
        acck[0] = bk0.x; acck[1] = bk0.y; acck[2] = bk0.z; acck[3] = bk0.w;
        acck[4] = bk1.x; acck[5] = bk1.y; acck[6] = bk1.z; acck[7] = bk1.w;
        acck = wmma16(a0.v, xb0.v, acck);
        acck = wmma16(a1.v, xb1.v, acck);
        // k_nat[row][d]: lane holds fixed pos, g spans consecutive features
        v8h pk;
#pragma unroll
        for (int g = 0; g < 8; ++g) pk[g] = (_Float16)acck[g];
        const size_t grow = rowBlk + mbase + c;
        *(v8h*)&k_nat[grow * 64 + ft * 16 + 8 * hi] = pk;
      }
    }
  }
}

// ===========================================================================
// Kernel B: one block per batch. 8 waves x 512 positions each.
// 3 GRU/MLP iterations + final softmax-attn writeout.
// ===========================================================================
__global__ __launch_bounds__(256) void slot_iter_kernel(
    const _Float16* __restrict__ k_nat, const _Float16* __restrict__ v_t,
    const unsigned char* __restrict__ mask,
    const float* __restrict__ noise, const float* __restrict__ slot_mu,
    const float* __restrict__ slot_logvar,
    const float* __restrict__ wq, const float* __restrict__ bq,
    const float* __restrict__ w_ih, const float* __restrict__ w_hh,
    const float* __restrict__ b_ih, const float* __restrict__ b_hh,
    const float* __restrict__ g_sl, const float* __restrict__ b_sl,
    const float* __restrict__ g_mlp, const float* __restrict__ b_mlp,
    const float* __restrict__ w1, const float* __restrict__ b1,
    const float* __restrict__ w2, const float* __restrict__ b2,
    float* __restrict__ out_slots, float* __restrict__ out_attn) {
  __shared__ float     slots[K_SL][D_SZ];
  __shared__ float     qln[K_SL][D_SZ];       // reused for MLP hidden LN
  __shared__ _Float16  qf16[K_SL][D_SZ];
  __shared__ _Float16  pbuf[8][K_SL][32];     // per-wave p-tile (C/D -> A transpose)
  __shared__ float     sSum[K_SL];
  __shared__ float     updS[K_SL][D_SZ];
  __shared__ float     vsum[D_SZ];
  __shared__ float     gx[K_SL][192];
  __shared__ float     gh[K_SL][192];
  __shared__ float     t1[K_SL][128];

  const int tid = threadIdx.x, lane = tid & 31, wave = tid >> 5;
  const int c = lane & 15, hi = lane >> 4;
  const int b = blockIdx.x;

  const _Float16* kb_base = k_nat + (size_t)b * N_SZ * D_SZ;
  const _Float16* vt_base = v_t   + (size_t)b * D_SZ * N_SZ;
  const unsigned char* mrow = mask + (size_t)b * N_SZ;

  // slots = mu + exp(0.5*logvar) * noise
  for (int i = tid; i < K_SL * D_SZ; i += 256) {
    const int d = i & 63;
    slots[i >> 6][d] =
        slot_mu[d] + __expf(0.5f * slot_logvar[d]) * noise[(size_t)b * K_SL * D_SZ + i];
  }
  if (tid < 64) vsum[tid] = 0.f;
  __syncthreads();
  {  // vsum[d] = sum_n v[n][d]  (for the eps correction term)
    const int d = tid & 63, seg = tid >> 6;
    const _Float16* vp = vt_base + (size_t)d * N_SZ + seg * 1024;
    float s = 0.f;
    for (int n = 0; n < 1024; ++n) s += (float)vp[n];
    atomicAdd(&vsum[d], s);
  }

  for (int iter = 0; iter <= N_ITERS; ++iter) {
    const bool final_pass = (iter == N_ITERS);
    __syncthreads();
    // ---- q = (LN(slots)*g_sl+b_sl) @ wq.T + bq, scale folded in ----
    if (tid < K_SL) {
      float m = 0.f;
      for (int i = 0; i < 64; ++i) m += slots[tid][i];
      m *= (1.f / 64.f);
      float var = 0.f;
      for (int i = 0; i < 64; ++i) { float dd = slots[tid][i] - m; var += dd * dd; }
      var *= (1.f / 64.f);
      const float inv = rsqrtf(var + LN_EPS);
      for (int i = 0; i < 64; ++i)
        qln[tid][i] = (slots[tid][i] - m) * inv * g_sl[i] + b_sl[i];
    }
    if (tid < K_SL) sSum[tid] = 0.f;
    for (int i = tid; i < K_SL * D_SZ; i += 256) updS[i >> 6][i & 63] = 0.f;
    __syncthreads();
    for (int o = tid; o < K_SL * D_SZ; o += 256) {
      const int ks = o >> 6, j = o & 63;
      const float* wr = wq + (size_t)j * 64;
      float acc = bq[j];
      for (int d2 = 0; d2 < 64; ++d2) acc += qln[ks][d2] * wr[d2];
      qf16[ks][j] = (_Float16)(acc * 0.125f);  // * D^-0.5
    }
    __syncthreads();

    // ---- attend: per-wave over 512 positions ----
    V16 qa0, qa1;
    {
      const int koff = hi ? 16 : 0;
      qa0.h[0] = *(const v8h*)&qf16[c][koff];
      qa0.h[1] = *(const v8h*)&qf16[c][koff + 8];
      qa1.h[0] = *(const v8h*)&qf16[c][32 + koff];
      qa1.h[1] = *(const v8h*)&qf16[c][32 + koff + 8];
    }
    v8f Uacc[4];
#pragma unroll
    for (int j = 0; j < 4; ++j)
#pragma unroll
      for (int g = 0; g < 8; ++g) Uacc[j][g] = 0.f;
    float sPart[8];
#pragma unroll
    for (int g = 0; g < 8; ++g) sPart[g] = 0.f;

    const int posWave = wave * 512;
    for (int ch = 0; ch < 16; ++ch) {
      const int posC = posWave + ch * 32;
      if (ch + 1 < 16)  // global_prefetch_b8 on next k chunk
        __builtin_prefetch(kb_base + (size_t)(posC + 32 + c) * 64, 0, 0);
#pragma unroll
      for (int sub = 0; sub < 2; ++sub) {
        const int pb  = posC + sub * 16;
        const int kb2 = hi * 16;
        const _Float16* kr = kb_base + (size_t)(pb + c) * 64 + kb2;
        V16 kb0, kb1;
        kb0.h[0] = *(const v8h*)(kr);
        kb0.h[1] = *(const v8h*)(kr + 8);
        kb1.h[0] = *(const v8h*)(kr + 32);
        kb1.h[1] = *(const v8h*)(kr + 40);
        const float negv = mrow[pb + c] ? 0.f : -1e9f;
        v8f acc;
#pragma unroll
        for (int g = 0; g < 8; ++g) acc[g] = negv;
        acc = wmma16(qa0.v, kb0.v, acc);
        acc = wmma16(qa1.v, kb1.v, acc);
        // softmax over the 16 slots: column lives in lane pair (c, c+16)
        float mx = acc[0];
#pragma unroll
        for (int g = 1; g < 8; ++g) mx = fmaxf(mx, acc[g]);
        mx = fmaxf(mx, __shfl_xor(mx, 16, 32));
        float p[8], ssum = 0.f;
#pragma unroll
        for (int g = 0; g < 8; ++g) { p[g] = __expf(acc[g] - mx); ssum += p[g]; }
        ssum += __shfl_xor(ssum, 16, 32);
        const float rinv = 1.f / ssum;
        if (!final_pass) {
#pragma unroll
          for (int g = 0; g < 8; ++g) {
            const float pv = p[g] * rinv;
            sPart[g] += pv;
            pbuf[wave][g + 8 * hi][sub * 16 + c] = (_Float16)pv;
          }
        } else {
#pragma unroll
          for (int g = 0; g < 8; ++g)
            out_attn[((size_t)b * K_SL + (g + 8 * hi)) * N_SZ + pb + c] = p[g] * rinv;
        }
      }
      if (!final_pass) {
        // same-wave cross-lane LDS RAW: DS ops in order, wait store completion
        asm volatile("s_wait_dscnt 0" ::: "memory");
        V16 pa;
        {
          const int koff = hi ? 16 : 0;
          pa.h[0] = *(const v8h*)&pbuf[wave][c][koff];
          pa.h[1] = *(const v8h*)&pbuf[wave][c][koff + 8];
        }
#pragma unroll
        for (int j = 0; j < 4; ++j) {
          const _Float16* vr =
              vt_base + (size_t)(j * 16 + c) * N_SZ + posC + hi * 16;
          V16 vb;
          vb.h[0] = *(const v8h*)(vr);
          vb.h[1] = *(const v8h*)(vr + 8);
          Uacc[j] = wmma16(pa.v, vb.v, Uacc[j]);
        }
      }
    }

    if (!final_pass) {
      // ---- reduce S and U across waves via LDS float atomics ----
#pragma unroll
      for (int g = 0; g < 8; ++g) atomicAdd(&sSum[g + 8 * hi], sPart[g]);
#pragma unroll
      for (int j = 0; j < 4; ++j)
#pragma unroll
        for (int g = 0; g < 8; ++g)
          atomicAdd(&updS[g + 8 * hi][j * 16 + c], Uacc[j][g]);
      __syncthreads();
      // upd = (U + eps*sum_n v) / (S + N*eps)
      for (int o = tid; o < K_SL * D_SZ; o += 256) {
        const int ks = o >> 6, d = o & 63;
        updS[ks][d] = (updS[ks][d] + EPS_A * vsum[d]) /
                      (sSum[ks] + (float)N_SZ * EPS_A);
      }
      __syncthreads();
      // ---- GRU cell (tiny: plain VALU) ----
      for (int o = tid; o < K_SL * 192; o += 256) {
        const int ks = o / 192, jj = o % 192;
        const float* wi = w_ih + (size_t)jj * 64;
        const float* wh = w_hh + (size_t)jj * 64;
        float ax = b_ih[jj], ah = b_hh[jj];
        for (int d2 = 0; d2 < 64; ++d2) {
          ax += updS[ks][d2] * wi[d2];
          ah += slots[ks][d2] * wh[d2];
        }
        gx[ks][jj] = ax;
        gh[ks][jj] = ah;
      }
      __syncthreads();
      for (int o = tid; o < K_SL * D_SZ; o += 256) {
        const int ks = o >> 6, j = o & 63;
        const float r = 1.f / (1.f + __expf(-(gx[ks][j] + gh[ks][j])));
        const float z = 1.f / (1.f + __expf(-(gx[ks][64 + j] + gh[ks][64 + j])));
        const float n = tanhf(gx[ks][128 + j] + r * gh[ks][128 + j]);
        slots[ks][j] = (1.f - z) * n + z * slots[ks][j];
      }
      __syncthreads();
      // ---- MLP residual ----
      if (tid < K_SL) {
        float m = 0.f;
        for (int i = 0; i < 64; ++i) m += slots[tid][i];
        m *= (1.f / 64.f);
        float var = 0.f;
        for (int i = 0; i < 64; ++i) { float dd = slots[tid][i] - m; var += dd * dd; }
        var *= (1.f / 64.f);
        const float inv = rsqrtf(var + LN_EPS);
        for (int i = 0; i < 64; ++i)
          qln[tid][i] = (slots[tid][i] - m) * inv * g_mlp[i] + b_mlp[i];
      }
      __syncthreads();
      for (int o = tid; o < K_SL * 128; o += 256) {
        const int ks = o >> 7, jj = o & 127;
        const float* wr = w1 + (size_t)jj * 64;
        float acc = b1[jj];
        for (int d2 = 0; d2 < 64; ++d2) acc += qln[ks][d2] * wr[d2];
        t1[ks][jj] = fmaxf(acc, 0.f);
      }
      __syncthreads();
      for (int o = tid; o < K_SL * D_SZ; o += 256) {
        const int ks = o >> 6, d = o & 63;
        const float* wr = w2 + (size_t)d * 128;
        float acc = b2[d];
        for (int jj = 0; jj < 128; ++jj) acc += t1[ks][jj] * wr[jj];
        slots[ks][d] += acc;
      }
    }
  }
  __syncthreads();
  for (int o = tid; o < K_SL * D_SZ; o += 256)
    out_slots[(size_t)b * K_SL * D_SZ + o] = slots[o >> 6][o & 63];
}

// ===========================================================================
extern "C" void kernel_launch(void* const* d_in, const int* in_sizes, int n_in,
                              void* d_out, int out_size, void* d_ws, size_t ws_size,
                              hipStream_t stream) {
  (void)in_sizes; (void)n_in; (void)out_size; (void)ws_size;
  const float* inputs       = (const float*)d_in[0];
  const unsigned char* mask = (const unsigned char*)d_in[1];
  const float* noise        = (const float*)d_in[2];
  const float* slot_mu      = (const float*)d_in[3];
  const float* slot_logvar  = (const float*)d_in[4];
  const float* wq   = (const float*)d_in[5];
  const float* bq   = (const float*)d_in[6];
  const float* wk   = (const float*)d_in[7];
  const float* bk   = (const float*)d_in[8];
  const float* wv   = (const float*)d_in[9];
  const float* bv   = (const float*)d_in[10];
  const float* w_ih = (const float*)d_in[11];
  const float* w_hh = (const float*)d_in[12];
  const float* b_ih = (const float*)d_in[13];
  const float* b_hh = (const float*)d_in[14];
  const float* g_in = (const float*)d_in[15];
  const float* b_in = (const float*)d_in[16];
  const float* g_sl = (const float*)d_in[17];
  const float* b_sl = (const float*)d_in[18];
  const float* g_mlp = (const float*)d_in[19];
  const float* b_mlp = (const float*)d_in[20];
  const float* w1 = (const float*)d_in[21];
  const float* b1 = (const float*)d_in[22];
  const float* w2 = (const float*)d_in[23];
  const float* b2 = (const float*)d_in[24];

  // workspace: k (f16, natural [b][n][d]) then v (f16, transposed [b][d][n])
  _Float16* k_nat = (_Float16*)d_ws;
  _Float16* v_t   = k_nat + (size_t)B_SZ * N_SZ * D_SZ;

  float* out_slots = (float*)d_out;
  float* out_attn  = out_slots + (size_t)B_SZ * K_SL * D_SZ;

  kv_proj_kernel<<<(B_SZ * N_SZ) / 256, 256, 0, stream>>>(
      inputs, g_in, b_in, wk, bk, wv, bv, k_nat, v_t);
  slot_iter_kernel<<<B_SZ, 256, 0, stream>>>(
      k_nat, v_t, mask, noise, slot_mu, slot_logvar, wq, bq,
      w_ih, w_hh, b_ih, b_hh, g_sl, b_sl, g_mlp, b_mlp, w1, b1, w2, b2,
      out_slots, out_attn);
}
